// MixtralDecoderLayer_53223234732074
// MI455X (gfx1250) — matrix operand order repaired and checked by reading, not compiled, not run on hardware
//
#include <hip/hip_runtime.h>
#include <hip/hip_bf16.h>
#include <math.h>

typedef __attribute__((ext_vector_type(16))) __bf16 v16bf;
typedef __attribute__((ext_vector_type(4)))  __bf16 v4bf;
typedef __attribute__((ext_vector_type(8)))  float  v8f;

#define S_LEN 1024
#define HDIM  2048
#define NHEAD 16
#define KVH   4
#define HD    128
#define NEXP  8
#define FDIM  4096

// --------------------------------------------------------- CDNA5 async copy
__device__ __forceinline__ void async_copy_b128(const void* gptr, void* lptr) {
  unsigned lds = (unsigned)(size_t)lptr;   // low 32 bits of generic shared ptr = LDS offset
  asm volatile("global_load_async_to_lds_b128 %0, %1, off"
               :: "v"(lds), "v"(gptr) : "memory");
}
__device__ __forceinline__ void async_wait0() {
  asm volatile("s_wait_asynccnt 0x0" ::: "memory");
}

// ---------------------------------------------------------------- rmsnorm
__global__ __launch_bounds__(256) void rmsnorm_kernel(
    const float* __restrict__ x, const float* __restrict__ w, float* __restrict__ y) {
  int row = blockIdx.x;
  const float* xr = x + (size_t)row * HDIM;
  float* yr = y + (size_t)row * HDIM;
  __shared__ float red[256];
  float v[8]; float s = 0.f;
#pragma unroll
  for (int i = 0; i < 8; ++i) { v[i] = xr[threadIdx.x + i * 256]; s += v[i] * v[i]; }
  red[threadIdx.x] = s; __syncthreads();
  for (int off = 128; off > 0; off >>= 1) {
    if (threadIdx.x < off) red[threadIdx.x] += red[threadIdx.x + off];
    __syncthreads();
  }
  float rs = rsqrtf(red[0] / (float)HDIM + 1e-6f);
#pragma unroll
  for (int i = 0; i < 8; ++i) {
    int c = threadIdx.x + i * 256;
    yr[c] = w[c] * (v[i] * rs);
  }
}

// ------------------------------------------------- generic 128x128 WMMA GEMM
// Double-buffered LDS, K-step 64, pipelined staging, 2m x 4n wave tiling.
// A: fp32 [M,K] (a_bf16=0, optional row-gather) or bf16 [M,K] (a_bf16=1, async-to-LDS)
// B: b_layout 0: fp32 [K,N] row-major (cvt+ds_store) | 2: bf16 [N,K] (async-to-LDS)
// modes: 0 store  1 +residual(R)  2 scatter C[gather[m]] += rowscale*acc
//        3 alpha*acc + causal/attn mask  4 Cbf = bf16(silu(R)*acc)
__global__ __launch_bounds__(256) void gemm128_kernel(
    const void* __restrict__ A, int lda, int a_bf16, long az_stride,
    const void* __restrict__ B, int ldb, int b_layout, long bz_stride, int bz_div,
    float* __restrict__ C, int ldc, long cz_stride,
    const float* __restrict__ R,
    __bf16* __restrict__ Cbf,
    const float* __restrict__ rowscale, int rs_stride,
    const unsigned char* __restrict__ amask,
    const int* __restrict__ gatherA,
    const int* __restrict__ cntp,
    int mode, float alpha, int K)
{
  const int m0 = blockIdx.y * 128, n0 = blockIdx.x * 128;
  int cap = 0x7fffffff;
  if (cntp) {
    cap = cntp[0];
    if (m0 >= cap) return;                 // uniform early-exit (before barriers)
  }

  __shared__ __align__(16) __bf16 lsA[2][128 * 64];   // [m][k]
  __shared__ __align__(16) __bf16 lsB[2][128 * 72];   // B^T: [n][k], stride 72 (pad 8)
  const int t = threadIdx.x;
  const int wave = t >> 5, lane = t & 31;
  const int half = lane >> 4, l16 = lane & 15;
  const int wrow = wave & 3, wcol = wave >> 2;        // 2m x 4n tiling
  const int z  = blockIdx.z;

  const size_t aoff = (size_t)z * (size_t)az_stride;
  const size_t boff = (size_t)(z / bz_div) * (size_t)bz_stride;
  float* Cz = C + (size_t)z * (size_t)cz_stride;
  const bool anyasync = a_bf16 || (b_layout == 2);

  float4 aS[8];   // staged fp32 A regs
  float4 bS[8];   // staged fp32 B regs
  v8f acc[2][4] = {};

  auto stage_issue = [&](int kb, int buf) {
    if (a_bf16) {
      const __bf16* Ab = (const __bf16*)A + aoff;
#pragma unroll
      for (int i = 0; i < 4; ++i) {
        int r = i * 32 + (t >> 3), c = (t & 7) * 8;
        async_copy_b128(Ab + (size_t)(m0 + r) * lda + kb + c, &lsA[buf][r * 64 + c]);
      }
    } else {
      const float* Af = (const float*)A + aoff;
#pragma unroll
      for (int i = 0; i < 8; ++i) {
        int r = i * 16 + (t >> 4), c = (t & 15) * 4;
        int row = m0 + r;
        if (gatherA) row = gatherA[row < cap ? row : (cap - 1)];
        aS[i] = *(const float4*)(Af + (size_t)row * lda + kb + c);
      }
    }
    if (b_layout == 2) {
      const __bf16* Bb = (const __bf16*)B + boff;
#pragma unroll
      for (int i = 0; i < 4; ++i) {
        int n = i * 32 + (t >> 3), k = (t & 7) * 8;
        async_copy_b128(Bb + (size_t)(n0 + n) * ldb + kb + k, &lsB[buf][n * 72 + k]);
      }
    } else {
      const float* Bf = (const float*)B + boff;
#pragma unroll
      for (int i = 0; i < 8; ++i) {
        int k = i * 8 + (t >> 5), n = (t & 31) * 4;
        bS[i] = *(const float4*)(Bf + (size_t)(kb + k) * ldb + n0 + n);
      }
    }
  };

  auto stage_commit = [&](int buf) {
    if (!a_bf16) {
#pragma unroll
      for (int i = 0; i < 8; ++i) {
        int r = i * 16 + (t >> 4), c = (t & 15) * 4;
        v4bf h; h.x = (__bf16)aS[i].x; h.y = (__bf16)aS[i].y;
                h.z = (__bf16)aS[i].z; h.w = (__bf16)aS[i].w;
        *(v4bf*)&lsA[buf][r * 64 + c] = h;
      }
    }
    if (b_layout != 2) {
#pragma unroll
      for (int i = 0; i < 8; ++i) {
        int k = i * 8 + (t >> 5), n = (t & 31) * 4;
        lsB[buf][(n + 0) * 72 + k] = (__bf16)bS[i].x;
        lsB[buf][(n + 1) * 72 + k] = (__bf16)bS[i].y;
        lsB[buf][(n + 2) * 72 + k] = (__bf16)bS[i].z;
        lsB[buf][(n + 3) * 72 + k] = (__bf16)bS[i].w;
      }
    }
  };

  // prologue: stage first tile
  stage_issue(0, 0);
  stage_commit(0);
  if (anyasync) async_wait0();
  __syncthreads();

  int cur = 0;
  for (int kb = 0; kb < K; kb += 64) {
    const bool more = (kb + 64) < K;
    if (more) stage_issue(kb + 64, cur ^ 1);   // global loads in flight during compute

    // ---- compute current buffer: 2 k-substeps x (2m x 4n) tiles
#pragma unroll
    for (int ks = 0; ks < 2; ++ks) {
      const int kk = ks * 32;
      const __bf16* pa0 = &lsA[cur][(wrow * 32 + l16) * 64 + kk];
      const __bf16* pa1 = pa0 + 16 * 64;
      v16bf af0, af1;
      ((uint4*)&af0)[0] = *(const uint4*)(pa0 + half * 8);
      ((uint4*)&af0)[1] = *(const uint4*)(pa0 + 16 + half * 8);
      ((uint4*)&af1)[0] = *(const uint4*)(pa1 + half * 8);
      ((uint4*)&af1)[1] = *(const uint4*)(pa1 + 16 + half * 8);
      v16bf bcur;
      {
        const __bf16* pb = &lsB[cur][((wcol * 4) * 16 + l16) * 72 + kk + half * 8];
        ((uint4*)&bcur)[0] = *(const uint4*)(pb);
        ((uint4*)&bcur)[1] = *(const uint4*)(pb + 16);
      }
#pragma unroll
      for (int ni = 0; ni < 4; ++ni) {
        v16bf bnxt;
        if (ni < 3) {                       // prefetch next B fragment
          const __bf16* pb = &lsB[cur][((wcol * 4 + ni + 1) * 16 + l16) * 72 + kk + half * 8];
          ((uint4*)&bnxt)[0] = *(const uint4*)(pb);
          ((uint4*)&bnxt)[1] = *(const uint4*)(pb + 16);
        }
        acc[0][ni] = __builtin_amdgcn_wmma_f32_16x16x32_bf16(
            false, af0, false, bcur, (short)0, acc[0][ni], false, false);
        acc[1][ni] = __builtin_amdgcn_wmma_f32_16x16x32_bf16(
            false, af1, false, bcur, (short)0, acc[1][ni], false, false);
        if (ni < 3) bcur = bnxt;
      }
    }

    if (more) {
      stage_commit(cur ^ 1);
      if (anyasync) async_wait0();
    }
    __syncthreads();
    cur ^= 1;
  }

  // ---- epilogue
#pragma unroll
  for (int mi = 0; mi < 2; ++mi) {
#pragma unroll
    for (int ni = 0; ni < 4; ++ni) {
      int gn = n0 + (wcol * 4 + ni) * 16 + l16;
#pragma unroll
      for (int p = 0; p < 8; ++p) {
        int gm = m0 + wrow * 32 + mi * 16 + half * 8 + p;
        bool valid = gm < cap;
        float v = acc[mi][ni][p] * alpha;
        size_t ci = (size_t)gm * ldc + gn;
        if (mode == 0) {
          if (valid) Cz[ci] = v;
        } else if (mode == 1) {
          Cz[ci] = R[ci] + v;
        } else if (mode == 2) {
          if (valid) {
            int row = gatherA ? gatherA[gm] : gm;
            C[(size_t)row * ldc + gn] += rowscale[(size_t)row * rs_stride] * v;
          }
        } else if (mode == 3) {
          bool ok = (gn <= gm) && (amask[gn] != 0);
          Cz[ci] = ok ? v : -3.0e38f;
        } else {  // mode 4: inter = silu(up) * gate, bf16
          if (valid) {
            float u = R[ci];
            Cbf[ci] = (__bf16)((u / (1.f + __expf(-u))) * v);
          }
        }
      }
    }
  }
}

// ---------------------------------------------------------------- RoPE -> bf16
__global__ void rope_kernel(const float* __restrict__ q, const float* __restrict__ k,
                            const float* __restrict__ cosb, const float* __restrict__ sinb,
                            __bf16* __restrict__ qo, __bf16* __restrict__ ko) {
  int s = blockIdx.x, h = blockIdx.y, d = threadIdx.x;
  float c = cosb[s * HD + d], si = sinb[s * HD + d];
  if (h < NHEAD) {
    const float* src = q + (size_t)s * (NHEAD * HD) + h * HD;
    float x = src[d];
    float xr = (d < HD / 2) ? -src[d + HD / 2] : src[d - HD / 2];
    qo[(size_t)s * (NHEAD * HD) + h * HD + d] = (__bf16)(x * c + xr * si);
  } else {
    int hk = h - NHEAD;
    const float* src = k + (size_t)s * (KVH * HD) + hk * HD;
    float x = src[d];
    float xr = (d < HD / 2) ? -src[d + HD / 2] : src[d - HD / 2];
    ko[(size_t)s * (KVH * HD) + hk * HD + d] = (__bf16)(x * c + xr * si);
  }
}

// ---------------------------------------- V -> V^T bf16 (LDS tile transpose)
__global__ __launch_bounds__(256) void v_transpose_kernel(const float* __restrict__ v,
                                                          __bf16* __restrict__ vt) {
  __shared__ float tile[64][65];
  int s0 = blockIdx.x * 64;           // token tile
  int d0 = blockIdx.y * 64;           // feature tile (KVH*HD = 512)
  int col = threadIdx.x & 63, row4 = threadIdx.x >> 6;
#pragma unroll
  for (int i = 0; i < 16; ++i) {
    int r = i * 4 + row4;
    tile[r][col] = v[(size_t)(s0 + r) * (KVH * HD) + d0 + col];
  }
  __syncthreads();
#pragma unroll
  for (int i = 0; i < 16; ++i) {
    int r = i * 4 + row4;
    vt[(size_t)(d0 + r) * S_LEN + s0 + col] = (__bf16)tile[col][r];
  }
}

// ---------------------------------------------------------------- softmax row
__global__ __launch_bounds__(256) void softmax_kernel(const float* __restrict__ sc,
                                                      __bf16* __restrict__ pr) {
  size_t base = (size_t)blockIdx.x * S_LEN;
  __shared__ float red[256];
  float v[4]; float mx = -3.4e38f;
#pragma unroll
  for (int i = 0; i < 4; ++i) { v[i] = sc[base + threadIdx.x + i * 256]; mx = fmaxf(mx, v[i]); }
  red[threadIdx.x] = mx; __syncthreads();
  for (int off = 128; off > 0; off >>= 1) {
    if (threadIdx.x < off) red[threadIdx.x] = fmaxf(red[threadIdx.x], red[threadIdx.x + off]);
    __syncthreads();
  }
  mx = red[0]; __syncthreads();
  float s = 0.f;
#pragma unroll
  for (int i = 0; i < 4; ++i) { v[i] = __expf(v[i] - mx); s += v[i]; }
  red[threadIdx.x] = s; __syncthreads();
  for (int off = 128; off > 0; off >>= 1) {
    if (threadIdx.x < off) red[threadIdx.x] += red[threadIdx.x + off];
    __syncthreads();
  }
  float inv = 1.0f / red[0];
#pragma unroll
  for (int i = 0; i < 4; ++i) pr[base + threadIdx.x + i * 256] = (__bf16)(v[i] * inv);
}

// ---------------------------------------------------------------- router top-2
__global__ void router_kernel(const float* __restrict__ xn, const float* __restrict__ rw,
                              float* __restrict__ wte) {
  int tkn = blockIdx.x * 64 + threadIdx.x;
  if (tkn >= S_LEN) return;
  float lg[NEXP] = {};
  const float* xr = xn + (size_t)tkn * HDIM;
  for (int h = 0; h < HDIM; ++h) {
    float xv = xr[h];
    const float* r = rw + h * NEXP;
#pragma unroll
    for (int e = 0; e < NEXP; ++e) lg[e] += xv * r[e];
  }
  float mx = lg[0];
#pragma unroll
  for (int e = 1; e < NEXP; ++e) mx = fmaxf(mx, lg[e]);
  float p[NEXP]; float s = 0.f;
#pragma unroll
  for (int e = 0; e < NEXP; ++e) { p[e] = __expf(lg[e] - mx); s += p[e]; }
#pragma unroll
  for (int e = 0; e < NEXP; ++e) p[e] /= s;
  int i1 = 0;
  for (int e = 1; e < NEXP; ++e) if (p[e] > p[i1]) i1 = e;     // first max (jax tie rule)
  int i2 = -1;
  for (int e = 0; e < NEXP; ++e) { if (e == i1) continue; if (i2 < 0 || p[e] > p[i2]) i2 = e; }
  float wsum = p[i1] + p[i2];
  float* o = wte + (size_t)tkn * NEXP;
#pragma unroll
  for (int e = 0; e < NEXP; ++e) o[e] = 0.f;
  o[i1] = p[i1] / wsum; o[i2] = p[i2] / wsum;
}

// ---------------------------- deterministic per-expert token lists (top-2)
__global__ void moe_gather_kernel(const float* __restrict__ wte,
                                  int* __restrict__ lists, int* __restrict__ counts) {
  int w = threadIdx.x >> 5;      // expert id, one wave32 per expert
  int lane = threadIdx.x & 31;
  int cnt = 0;
  for (int base = 0; base < S_LEN; base += 32) {
    int tkn = base + lane;
    bool sel = wte[(size_t)tkn * NEXP + w] > 0.f;
    unsigned mask = (unsigned)__ballot(sel);
    int pre = __popc(mask & ((1u << lane) - 1u));
    if (sel) lists[w * S_LEN + cnt + pre] = tkn;
    cnt += __popc(mask);
  }
  if (lane == 0) counts[w] = cnt;
}

// ================================================================ launch
extern "C" void kernel_launch(void* const* d_in, const int* in_sizes, int n_in,
                              void* d_out, int out_size, void* d_ws, size_t ws_size,
                              hipStream_t stream) {
  const float* hidden = (const float*)d_in[0];
  const unsigned char* amask = (const unsigned char*)d_in[1];
  const float* cosb = (const float*)d_in[2];
  const float* sinb = (const float*)d_in[3];
  // d_in[4..5]: zero KV caches — masked out (CACHE_IDX=0, pad < S), unused.
  const float* q_w = (const float*)d_in[6];
  const float* k_w = (const float*)d_in[7];
  const float* v_w = (const float*)d_in[8];
  const float* o_w = (const float*)d_in[9];
  const float* ln1 = (const float*)d_in[10];
  const float* ln2 = (const float*)d_in[11];
  const float* router_w = (const float*)d_in[12];
  const float* up_w = (const float*)d_in[13];
  const float* gate_w = (const float*)d_in[14];
  const float* down_w = (const float*)d_in[15];
  float* out = (float*)d_out;
  (void)in_sizes; (void)n_in; (void)out_size; (void)ws_size;

  char* ws = (char*)d_ws;
  size_t off = 0;
  auto alloc = [&](size_t bytes) -> void* {
    void* p = ws + off; off += (bytes + 255) & ~(size_t)255; return p;
  };
  float*  xn1    = (float*) alloc(sizeof(float) * S_LEN * HDIM);
  float*  qbuf   = (float*) alloc(sizeof(float) * S_LEN * NHEAD * HD);
  float*  kbuf   = (float*) alloc(sizeof(float) * S_LEN * KVH * HD);
  float*  vbuf   = (float*) alloc(sizeof(float) * S_LEN * KVH * HD);
  __bf16* qbf    = (__bf16*)alloc(2ull * S_LEN * NHEAD * HD);
  __bf16* kbf    = (__bf16*)alloc(2ull * S_LEN * KVH * HD);
  __bf16* vt     = (__bf16*)alloc(2ull * KVH * HD * S_LEN);   // V^T [kv*HD, S]
  float*  scores = (float*) alloc(sizeof(float) * NHEAD * S_LEN * S_LEN);
  __bf16* probs  = (__bf16*)alloc(2ull * NHEAD * S_LEN * S_LEN);
  float*  attnb  = (float*) alloc(sizeof(float) * S_LEN * NHEAD * HD);
  float*  x1     = (float*) alloc(sizeof(float) * S_LEN * HDIM);
  float*  xn2    = (float*) alloc(sizeof(float) * S_LEN * HDIM);
  float*  wte    = (float*) alloc(sizeof(float) * S_LEN * NEXP);
  int*    lists  = (int*)   alloc(sizeof(int) * NEXP * S_LEN);
  int*    counts = (int*)   alloc(sizeof(int) * NEXP);
  float*  upb    = (float*) alloc(sizeof(float) * S_LEN * FDIM);
  __bf16* intb   = (__bf16*)alloc(2ull * S_LEN * FDIM);

  dim3 blk(256);
  const float inv_sqrt_hd = 0.08838834764831845f;  // 128^-0.5

  // 1) RMSNorm #1
  rmsnorm_kernel<<<S_LEN, blk, 0, stream>>>(hidden, ln1, xn1);

  // 2) Q / K / V projections
  gemm128_kernel<<<dim3(HDIM / 128, S_LEN / 128, 1), blk, 0, stream>>>(
      xn1, HDIM, 0, 0, q_w, NHEAD * HD, 0, 0, 1, qbuf, NHEAD * HD, 0,
      nullptr, nullptr, nullptr, 0, nullptr, nullptr, nullptr, 0, 1.0f, HDIM);
  gemm128_kernel<<<dim3((KVH * HD) / 128, S_LEN / 128, 1), blk, 0, stream>>>(
      xn1, HDIM, 0, 0, k_w, KVH * HD, 0, 0, 1, kbuf, KVH * HD, 0,
      nullptr, nullptr, nullptr, 0, nullptr, nullptr, nullptr, 0, 1.0f, HDIM);
  gemm128_kernel<<<dim3((KVH * HD) / 128, S_LEN / 128, 1), blk, 0, stream>>>(
      xn1, HDIM, 0, 0, v_w, KVH * HD, 0, 0, 1, vbuf, KVH * HD, 0,
      nullptr, nullptr, nullptr, 0, nullptr, nullptr, nullptr, 0, 1.0f, HDIM);

  // 3) RoPE(q,k) -> bf16 ; V -> V^T bf16
  rope_kernel<<<dim3(S_LEN, NHEAD + KVH), 128, 0, stream>>>(qbuf, kbuf, cosb, sinb, qbf, kbf);
  v_transpose_kernel<<<dim3(S_LEN / 64, (KVH * HD) / 64), blk, 0, stream>>>(vbuf, vt);

  // 4) scores[h] = alpha * Q_h @ K_{h/4}^T with causal+attention mask
  gemm128_kernel<<<dim3(S_LEN / 128, S_LEN / 128, NHEAD), blk, 0, stream>>>(
      qbf, NHEAD * HD, 1, HD, kbf, KVH * HD, 2, HD, 4,
      scores, S_LEN, (long)S_LEN * S_LEN,
      nullptr, nullptr, nullptr, 0, amask, nullptr, nullptr, 3, inv_sqrt_hd, HD);

  // 5) row softmax -> bf16 probs
  softmax_kernel<<<NHEAD * S_LEN, blk, 0, stream>>>(scores, probs);

  // 6) attn[h] = P_h @ V_{h/4} via V^T
  gemm128_kernel<<<dim3(HD / 128, S_LEN / 128, NHEAD), blk, 0, stream>>>(
      probs, S_LEN, 1, (long)S_LEN * S_LEN, vt, S_LEN, 2, (long)HD * S_LEN, 4,
      attnb, NHEAD * HD, HD,
      nullptr, nullptr, nullptr, 0, nullptr, nullptr, nullptr, 0, 1.0f, S_LEN);

  // 7) x1 = hidden + attn @ o_w
  gemm128_kernel<<<dim3(HDIM / 128, S_LEN / 128, 1), blk, 0, stream>>>(
      attnb, NHEAD * HD, 0, 0, o_w, HDIM, 0, 0, 1, x1, HDIM, 0,
      hidden, nullptr, nullptr, 0, nullptr, nullptr, nullptr, 1, 1.0f, NHEAD * HD);

  // 8) RMSNorm #2, router top-2, expert token lists
  rmsnorm_kernel<<<S_LEN, blk, 0, stream>>>(x1, ln2, xn2);
  router_kernel<<<S_LEN / 64, 64, 0, stream>>>(xn2, router_w, wte);
  moe_gather_kernel<<<1, 256, 0, stream>>>(wte, lists, counts);

  // 9) out = residual; experts accumulate weighted down-proj (top-2 sparse)
  hipMemcpyAsync(out, x1, sizeof(float) * S_LEN * HDIM, hipMemcpyDeviceToDevice, stream);

  for (int e = 0; e < NEXP; ++e) {
    const float* upw = up_w   + (size_t)e * HDIM * FDIM;
    const float* gtw = gate_w + (size_t)e * HDIM * FDIM;
    const float* dnw = down_w + (size_t)e * FDIM * HDIM;
    const int*   lst = lists + e * S_LEN;
    const int*   cnt = counts + e;
    // up (gathered rows of xn2)
    gemm128_kernel<<<dim3(FDIM / 128, S_LEN / 128, 1), blk, 0, stream>>>(
        xn2, HDIM, 0, 0, upw, FDIM, 0, 0, 1, upb, FDIM, 0,
        nullptr, nullptr, nullptr, 0, nullptr, lst, cnt, 0, 1.0f, HDIM);
    // gate (gathered) with fused silu(up)*gate -> bf16 inter
    gemm128_kernel<<<dim3(FDIM / 128, S_LEN / 128, 1), blk, 0, stream>>>(
        xn2, HDIM, 0, 0, gtw, FDIM, 0, 0, 1, upb, FDIM, 0,
        upb, intb, nullptr, 0, nullptr, lst, cnt, 4, 1.0f, HDIM);
    // down (compact bf16 A, async), scatter-accumulate weighted rows into out
    gemm128_kernel<<<dim3(HDIM / 128, S_LEN / 128, 1), blk, 0, stream>>>(
        intb, FDIM, 1, 0, dnw, HDIM, 0, 0, 1, out, HDIM, 0,
        nullptr, nullptr, wte + e, NEXP, nullptr, lst, cnt, 2, 1.0f, FDIM);
  }
}
